// TopEnsemble_48077863911971
// MI455X (gfx1250) — matrix operand order repaired and checked by reading, not compiled, not run on hardware
//
#include <hip/hip_runtime.h>
#include <cstdint>

// Problem sizes (from reference)
constexpr int NB = 1024;   // batch
constexpr int NS = 256;    // sequence (GEMM M per sample)
constexpr int ND = 512;    // model dim (GEMM K)
constexpr int NE = 16;     // experts
constexpr int NO = 512;    // out dim (GEMM N)

typedef __attribute__((ext_vector_type(16))) __bf16 bf16x16;
typedef __attribute__((ext_vector_type(8)))  __bf16 bf16x8;
typedef __attribute__((ext_vector_type(4)))  __bf16 bf16x4;
typedef __attribute__((ext_vector_type(8)))  float  f32x8;
typedef __attribute__((ext_vector_type(2)))  float  f32x2;

union FragAB { bf16x16 v; bf16x8 h[2]; bf16x4 q[4]; };

// CDNA5 async global->LDS copy (per-lane 16B), tracked by ASYNCcnt.
__device__ __forceinline__ void async_copy_b128(uint32_t lds_off, const void* gaddr) {
  asm volatile("global_load_async_to_lds_b128 %0, %1, off"
               :: "v"(lds_off), "v"(gaddr) : "memory");
}
#define WAIT_ASYNCCNT(N) asm volatile("s_wait_asynccnt %0" :: "n"(N) : "memory")

// ---------------------------------------------------------------------------
// Kernel 1: per-sample mean over S, gating linear (D x E), softmax, argmax.
// One block (8 wave32) per sample. Writes idx[b], gate[b].
// ---------------------------------------------------------------------------
__global__ __launch_bounds__(256) void gate_kernel(const float* __restrict__ x,
                                                   const float* __restrict__ Wg,
                                                   const float* __restrict__ bg,
                                                   int* __restrict__ idx,
                                                   float* __restrict__ gate)
{
  __shared__ float meanv[ND];
  __shared__ float sc[NE];
  const int b = blockIdx.x, t = threadIdx.x;
  const f32x2* xr = (const f32x2*)(x + (size_t)b * NS * ND);
  float s0 = 0.f, s1 = 0.f;
  for (int s = 0; s < NS; ++s) {
    f32x2 v = __builtin_nontemporal_load(&xr[(size_t)s * (ND / 2) + t]);
    s0 += v.x; s1 += v.y;
  }
  meanv[2 * t]     = s0 * (1.f / NS);
  meanv[2 * t + 1] = s1 * (1.f / NS);
  __syncthreads();
  const int wave = t >> 5, lane = t & 31;
  for (int e = wave; e < NE; e += 8) {
    float acc = 0.f;
    for (int d = lane; d < ND; d += 32) acc += meanv[d] * Wg[d * NE + e];
    for (int off = 16; off > 0; off >>= 1) acc += __shfl_down(acc, off, 32);
    if (lane == 0) sc[e] = acc + bg[e];
  }
  __syncthreads();
  if (t == 0) {
    float m = sc[0]; int bi = 0;
    for (int e = 1; e < NE; ++e) if (sc[e] > m) { m = sc[e]; bi = e; }
    float den = 0.f;
    for (int e = 0; e < NE; ++e) den += __expf(sc[e] - m);
    idx[b]  = bi;
    gate[b] = 1.f / den;   // softmax probability at the argmax
  }
}

// ---------------------------------------------------------------------------
// Kernel 2: We [E, D, O] f32  ->  WeT [E, O, D] bf16 (K-contiguous B operand)
// ---------------------------------------------------------------------------
__global__ __launch_bounds__(256) void wconv_kernel(const float* __restrict__ We,
                                                    __bf16* __restrict__ WeT)
{
  __shared__ float tile[32][33];
  const int e = blockIdx.z;
  const int d0 = blockIdx.y * 32, o0 = blockIdx.x * 32;
  const int tx = threadIdx.x, ty = threadIdx.y;
  const float* src = We + (size_t)e * ND * NO;
  for (int i = ty; i < 32; i += 8)
    tile[i][tx] = src[(size_t)(d0 + i) * NO + o0 + tx];
  __syncthreads();
  __bf16* dst = WeT + (size_t)e * NO * ND;
  for (int i = ty; i < 32; i += 8)
    dst[(size_t)(o0 + i) * ND + d0 + tx] = (__bf16)tile[tx][i];
}

// ---------------------------------------------------------------------------
// Kernel 3: per-sample GEMM  out[b] = gate * (x[b] @ W[idx[b]] + bias)
// Block tile 128(M) x 128(N), BK=32, 8 waves in a 4x2 grid, wave tile 32x64.
// Double-buffered LDS filled with global_load_async_to_lds_b128, pipelined
// with s_wait_asynccnt; compute = v_wmma_f32_16x16x32_bf16.
// ---------------------------------------------------------------------------
constexpr int BM = 128, BN = 128, BK = 32;
constexpr int AKP = BK + 4;   // fp32 A row pitch (pad 16B): 36 floats
constexpr int BKP = BK + 8;   // bf16 B row pitch (pad 16B): 40 bf16

__global__ __launch_bounds__(256) void moe_wmma_kernel(const float* __restrict__ x,
                                                       const __bf16* __restrict__ WeT,
                                                       const float* __restrict__ be,
                                                       const int* __restrict__ idx,
                                                       const float* __restrict__ gate,
                                                       float* __restrict__ out)
{
  __shared__ float  As[2][BM][AKP];   // 36 KB: A tile staged as fp32
  __shared__ __bf16 Bs[2][BN][BKP];   // 20 KB: B tile (N-major) as bf16

  const int b     = blockIdx.y;
  const int sTile = blockIdx.x >> 2;  // 2 M-tiles
  const int oTile = blockIdx.x & 3;   // 4 N-tiles
  const int t     = threadIdx.x;
  const int wave  = t >> 5, lane = t & 31;
  const int hf    = lane >> 4, r = lane & 15;
  const int wm    = (wave >> 1) * 32;
  const int wn    = (wave & 1) * 64;

  const int   ex = idx[b];
  const float g  = gate[b];

  const float*  Ag = x   + ((size_t)b  * NS + sTile * BM) * ND;
  const __bf16* Bg = WeT + ((size_t)ex * NO + oTile * BN) * ND;

  auto stage = [&](int buf, int k0) {
#pragma unroll
    for (int c = 0; c < 4; ++c) {              // A: 128x32 f32 = 1024 x 16B
      int chunk = t + c * 256;
      int row = chunk >> 3, c4 = chunk & 7;
      async_copy_b128((uint32_t)(uintptr_t)&As[buf][row][c4 * 4],
                      Ag + (size_t)row * ND + k0 + c4 * 4);
    }
#pragma unroll
    for (int c = 0; c < 2; ++c) {              // B: 128x32 bf16 = 512 x 16B
      int chunk = t + c * 256;
      int row = chunk >> 2, c8 = chunk & 3;
      async_copy_b128((uint32_t)(uintptr_t)&Bs[buf][row][c8 * 8],
                      Bg + (size_t)row * ND + k0 + c8 * 8);
    }
  };

  f32x8 acc[2][4];
  {
    f32x8 z = {0.f, 0.f, 0.f, 0.f, 0.f, 0.f, 0.f, 0.f};
#pragma unroll
    for (int mi = 0; mi < 2; ++mi)
#pragma unroll
      for (int ni = 0; ni < 4; ++ni) acc[mi][ni] = z;
  }

  stage(0, 0);                                  // prologue prefetch

  for (int it = 0; it < ND / BK; ++it) {
    if (it + 1 < ND / BK) {
      stage((it + 1) & 1, (it + 1) * BK);       // prefetch next buffer
      WAIT_ASYNCCNT(6);                         // current buffer's 6 ops done
    } else {
      WAIT_ASYNCCNT(0);
    }
    __syncthreads();

    const int buf = it & 1;
    // A fragments (16x32 bf16): lane half selects K {0..7,16..23}/{8..15,24..31}
    FragAB a[2];
#pragma unroll
    for (int mi = 0; mi < 2; ++mi) {
      const float* ap = &As[buf][wm + mi * 16 + r][hf * 8];
      float4 f0 = *(const float4*)(ap);
      float4 f1 = *(const float4*)(ap + 4);
      float4 f2 = *(const float4*)(ap + 16);
      float4 f3 = *(const float4*)(ap + 20);
      bf16x4 q0 = {(__bf16)f0.x, (__bf16)f0.y, (__bf16)f0.z, (__bf16)f0.w};
      bf16x4 q1 = {(__bf16)f1.x, (__bf16)f1.y, (__bf16)f1.z, (__bf16)f1.w};
      bf16x4 q2 = {(__bf16)f2.x, (__bf16)f2.y, (__bf16)f2.z, (__bf16)f2.w};
      bf16x4 q3 = {(__bf16)f3.x, (__bf16)f3.y, (__bf16)f3.z, (__bf16)f3.w};
      a[mi].q[0] = q0; a[mi].q[1] = q1; a[mi].q[2] = q2; a[mi].q[3] = q3;
    }
    // B fragments (32x16 bf16): lane = column, lane half selects K {0..7,16..23}/{8..15,24..31}
#pragma unroll
    for (int ni = 0; ni < 4; ++ni) {
      FragAB bb;
      const __bf16* bp = &Bs[buf][wn + ni * 16 + r][hf * 8];
      bb.h[0] = *(const bf16x8*)(bp);
      bb.h[1] = *(const bf16x8*)(bp + 16);
#pragma unroll
      for (int mi = 0; mi < 2; ++mi)
        acc[mi][ni] = __builtin_amdgcn_wmma_f32_16x16x32_bf16(
            false, a[mi].v, false, bb.v, (short)0, acc[mi][ni], false, false);
    }
    __syncthreads();
  }

  // Epilogue: out = gate * (acc + bias), streaming (nontemporal) stores.
  const float* bias = be + (size_t)ex * NO + oTile * BN;
#pragma unroll
  for (int ni = 0; ni < 4; ++ni) {
    const int n = wn + ni * 16 + r;
    const float bv = bias[n];
#pragma unroll
    for (int mi = 0; mi < 2; ++mi) {
      const int m0 = sTile * BM + wm + mi * 16 + hf * 8;
      float* op = out + ((size_t)b * NS + m0) * NO + oTile * BN + n;
#pragma unroll
      for (int j = 0; j < 8; ++j)
        __builtin_nontemporal_store(g * (acc[mi][ni][j] + bv), op + (size_t)j * NO);
    }
  }
}

// ---------------------------------------------------------------------------
extern "C" void kernel_launch(void* const* d_in, const int* in_sizes, int n_in,
                              void* d_out, int out_size, void* d_ws, size_t ws_size,
                              hipStream_t stream) {
  (void)in_sizes; (void)n_in; (void)out_size; (void)ws_size;
  const float* x  = (const float*)d_in[0];
  const float* Wg = (const float*)d_in[1];
  const float* bg = (const float*)d_in[2];
  const float* We = (const float*)d_in[3];
  const float* be = (const float*)d_in[4];
  float* out = (float*)d_out;

  char* ws = (char*)d_ws;
  __bf16* WeT = (__bf16*)ws;                               // 8 MB bf16 W^T
  size_t weBytes = (size_t)NE * NO * ND * sizeof(__bf16);
  int*   idx  = (int*)(ws + weBytes);
  float* gate = (float*)(ws + weBytes + (size_t)NB * sizeof(int));

  gate_kernel<<<NB, 256, 0, stream>>>(x, Wg, bg, idx, gate);
  wconv_kernel<<<dim3(NO / 32, ND / 32, NE), dim3(32, 8), 0, stream>>>(We, WeT);
  moe_wmma_kernel<<<dim3((NS / BM) * (NO / BN), NB), 256, 0, stream>>>(
      x, WeT, be, idx, gate, out);
}
